// TreeTLSTMQFunction_21887153341100
// MI455X (gfx1250) — matrix operand order repaired and checked by reading, not compiled, not run on hardware
//
#include <hip/hip_runtime.h>
#include <stdint.h>

typedef __bf16 bf16;
typedef __attribute__((ext_vector_type(16))) __bf16 v16bf;
typedef __attribute__((ext_vector_type(8)))  float  v8f;
typedef __attribute__((ext_vector_type(4)))  unsigned int u32x4;

#define DEV __device__ __forceinline__

union FragU { v16bf v; u32x4 q[2]; bf16 e[16]; };
union PackU { u32x4 q; bf16 e[8]; };

DEV v8f v8f_zero() { v8f z = {0.f,0.f,0.f,0.f,0.f,0.f,0.f,0.f}; return z; }

// A fragment (16x32 bf16, row-major source): lane holds row M=lane&15.
// lanes 0-15: K = k0 + {0..7, 16..23}; lanes 16-31: K = k0 + {8..15, 24..31}
DEV v16bf load_frag_a(const bf16* rowbase, int k0, int lane) {
  const bf16* p = rowbase + k0 + ((lane >> 4) << 3);
  FragU f;
  f.q[0] = *(const u32x4*)(p);
  f.q[1] = *(const u32x4*)(p + 16);
  return f.v;
}
// B fragment (32x16 bf16) from (N,K)-row-major weights: lane holds col N=lane&15,
// lanes 0-15: K = k0 + {0..7, 16..23} ; lanes 16-31: K = k0 + {8..15, 24..31}
DEV v16bf load_frag_b(const bf16* colbase, int k0, int lane) {
  const bf16* p = colbase + k0 + ((lane >> 4) << 3);
  FragU f;
  f.q[0] = *(const u32x4*)(p);
  f.q[1] = *(const u32x4*)(p + 16);
  return f.v;
}
DEV v8f wmma_bf16(v16bf a, v16bf b, v8f c) {
  return __builtin_amdgcn_wmma_f32_16x16x32_bf16(false, a, false, b, (short)0, c, false, false);
}
DEV float sigm(float x) { return 1.0f / (1.0f + __expf(-x)); }

// ---------------- prep kernels ----------------
__global__ void k_zero32(uint32_t* p, long n) {
  for (long i = blockIdx.x * (long)blockDim.x + threadIdx.x; i < n; i += (long)gridDim.x * blockDim.x) p[i] = 0u;
}
__global__ void k_cvt_bf16(const float* __restrict__ s, bf16* __restrict__ d, long n) {
  for (long i = blockIdx.x * (long)blockDim.x + threadIdx.x; i < n; i += (long)gridDim.x * blockDim.x) d[i] = (bf16)s[i];
}
// pad columns cs -> cd with zeros, convert to bf16
__global__ void k_padmat(const float* __restrict__ s, bf16* __restrict__ d, int cs, int cd, long n) {
  for (long i = blockIdx.x * (long)blockDim.x + threadIdx.x; i < n; i += (long)gridDim.x * blockDim.x) {
    long r = i / cd; int c = (int)(i - r * cd);
    d[i] = (c < cs) ? (bf16)s[r * cs + c] : (bf16)0.f;
  }
}
// src (R,C) f32 row-major -> dst (C,R) bf16 row-major
__global__ void k_transpose_bf16(const float* __restrict__ s, bf16* __restrict__ d, int R, int C, long n) {
  for (long i = blockIdx.x * (long)blockDim.x + threadIdx.x; i < n; i += (long)gridDim.x * blockDim.x) {
    long c = i / R; int r = (int)(i - c * R);
    d[i] = (bf16)s[(long)r * C + c];
  }
}
// adj (B,108,108) f32 -> adjbf (B,112,128) bf16 zero-padded
__global__ void k_adjprep(const float* __restrict__ adj, bf16* __restrict__ out, long n) {
  for (long i = blockIdx.x * (long)blockDim.x + threadIdx.x; i < n; i += (long)gridDim.x * blockDim.x) {
    long b = i / (112 * 128); long rem = i - b * (112 * 128);
    int r = (int)(rem >> 7); int k = (int)(rem & 127);
    float v = (r < 108 && k < 108) ? adj[((long)b * 108 + r) * 108 + k] : 0.f;
    out[i] = (bf16)v;
  }
}
// u[b,i] = sum_j adj[b,i,j] * jv[b,j]   (u padded to 112 rows, zeros beyond 108)
__global__ __launch_bounds__(128) void k_u(const float* __restrict__ adj, const float* __restrict__ jv, float* __restrict__ u) {
  int b = blockIdx.x, i = threadIdx.x;
  if (i < 112) {
    float s = 0.f;
    if (i < 108) {
      const float* row = adj + ((long)b * 108 + i) * 108;
      const float* v = jv + (long)b * 108;
      for (int j = 0; j < 108; j++) s += row[j] * v[j];
    }
    u[b * 112 + i] = s;
  }
}
// embedding gather + LayerNorm -> bf16. blocks [0,8192): leaves, [8192,8704): actions
__global__ __launch_bounds__(256)
void k_embed_ln(const int* __restrict__ leaf, const int* __restrict__ act,
                const float* __restrict__ emb, const float* __restrict__ g,
                const float* __restrict__ bta, bf16* __restrict__ hA, bf16* __restrict__ actenc) {
  const int x = blockIdx.x, t = threadIdx.x;
  int tid; bf16* dst;
  if (x < 512 * 16) { tid = leaf[x]; dst = hA + (long)x * 256; }
  else { int b = x - 512 * 16; tid = act[b]; dst = actenc + (long)b * 256; }
  float v = emb[(long)tid * 256 + t];
  __shared__ float red[8];
  float s = v;
  #pragma unroll
  for (int o = 16; o > 0; o >>= 1) s += __shfl_down(s, o, 32);
  if ((t & 31) == 0) red[t >> 5] = s;
  __syncthreads();
  float mean = 0.f;
  #pragma unroll
  for (int w = 0; w < 8; w++) mean += red[w];
  mean *= (1.f / 256.f);
  __syncthreads();
  float d = v - mean;
  float q = d * d;
  #pragma unroll
  for (int o = 16; o > 0; o >>= 1) q += __shfl_down(q, o, 32);
  if ((t & 31) == 0) red[t >> 5] = q;
  __syncthreads();
  float var = 0.f;
  #pragma unroll
  for (int w = 0; w < 8; w++) var += red[w];
  var *= (1.f / 256.f);
  dst[t] = (bf16)(d * rsqrtf(var + 1e-5f) * g[t] + bta[t]);
}

// ---------------- generic GEMM: out(M,256) = A(M,K)bf16 @ W(256,K)bf16^T + bias ----------------
// Waves in a block share the same s-tile (identical weight stream -> WGP$ hit), distinct m-tiles.
__global__ __launch_bounds__(256)
void k_gemm256(const bf16* __restrict__ A, const bf16* __restrict__ W,
               const float* __restrict__ bias, float* __restrict__ outF,
               bf16* __restrict__ outH, int K, int relu, int tiles_m) {
  const int lane = threadIdx.x & 31;
  const int wave = blockIdx.x * 8 + (threadIdx.x >> 5);
  const int st = wave / tiles_m;
  const int mt = wave - st * tiles_m;
  const int m0 = mt << 4;
  const int s0 = st << 4;
  const bf16* arow = A + (long)(m0 + (lane & 15)) * K;
  const bf16* wcol = W + (long)(s0 + (lane & 15)) * K;
  v8f acc = v8f_zero();
  for (int kk = 0; kk < K; kk += 32) {
    __builtin_prefetch(wcol + kk + 128, 0, 1);
    v16bf a = load_frag_a(arow, kk, lane);
    v16bf b = load_frag_b(wcol, kk, lane);
    acc = wmma_bf16(a, b, acc);
  }
  const int s = s0 + (lane & 15);
  const float bv = bias[s];
  const int hi = (lane >> 4) << 3;
  #pragma unroll
  for (int r = 0; r < 8; r++) {
    float v = acc[r] + bv;
    if (relu) v = fmaxf(v, 0.f);
    long idx = (long)(m0 + hi + r) * 256 + s;
    if (outF) outF[idx] = v; else outH[idx] = (bf16)v;
  }
}

// ---------------- Tree-LSTM level: fused 3 GEMMs (K=256) x 5 gates + LSTM epilogue ----------------
// Waves in a block share the same s-tile: all 15 weight-fragment streams (5 gates x {fc1,fc2,fc0})
// are identical across the block's 8 waves and stay resident in WGP$.
__global__ __launch_bounds__(256)
void k_tree_level(const bf16* __restrict__ hprev, const float* __restrict__ cprev,
                  const bf16* __restrict__ xin,
                  const bf16* __restrict__ wFC1, const bf16* __restrict__ wFC2, const bf16* __restrict__ wFC0,
                  const float* __restrict__ b1, const float* __restrict__ b2, const float* __restrict__ b0,
                  bf16* __restrict__ hout, float* __restrict__ cout,
                  int n, int log2n, int off, int tiles_m) {
  const int lane = threadIdx.x & 31;
  const int wave = blockIdx.x * 8 + (threadIdx.x >> 5);
  const int st = wave / tiles_m;
  const int mt = wave - st * tiles_m;
  const int m0 = mt << 4;                    // output row tile (over B*n)
  const int s0 = st << 4;                    // feature tile (over 256)
  const int mrow = m0 + (lane & 15);
  const int bL = mrow >> log2n;
  const int jL = mrow & (n - 1);
  const bf16* lhp = hprev + (long)(((bL << log2n) << 1) + (jL << 1)) * 256;
  const bf16* rhp = lhp + 256;
  const bf16* xp  = xin + (long)(bL * 15 + off + jL) * 256;
  const int coll = s0 + (lane & 15);
  v8f acc[5];
  #pragma unroll
  for (int g = 0; g < 5; g++) acc[g] = v8f_zero();
  for (int kk = 0; kk < 256; kk += 32) {
    v16bf aL = load_frag_a(lhp, kk, lane);
    v16bf aR = load_frag_a(rhp, kk, lane);
    v16bf aX = load_frag_a(xp,  kk, lane);
    #pragma unroll
    for (int g = 0; g < 5; g++) {
      const int col = g * 256 + coll;
      __builtin_prefetch(wFC1 + (long)col * 256 + kk + 64, 0, 1);
      v16bf bw;
      bw = load_frag_b(wFC1 + (long)col * 256, kk, lane); acc[g] = wmma_bf16(aL, bw, acc[g]);
      bw = load_frag_b(wFC2 + (long)col * 256, kk, lane); acc[g] = wmma_bf16(aR, bw, acc[g]);
      bw = load_frag_b(wFC0 + (long)col * 256, kk, lane); acc[g] = wmma_bf16(aX, bw, acc[g]);
    }
  }
  float bias[5];
  #pragma unroll
  for (int g = 0; g < 5; g++) { int c = g * 256 + coll; bias[g] = b0[c] + b1[c] + b2[c]; }
  const int hi = (lane >> 4) << 3;
  #pragma unroll
  for (int r = 0; r < 8; r++) {
    const int m = m0 + hi + r;
    const int bb = m >> log2n, jj = m & (n - 1);
    float av = acc[0][r] + bias[0];
    float iv = acc[1][r] + bias[1];
    float f1 = acc[2][r] + bias[2];
    float f2 = acc[3][r] + bias[3];
    float ov = acc[4][r] + bias[4];
    float lc = 0.f, rc = 0.f;
    if (cprev) {
      long cb = (long)(((bb << log2n) << 1) + (jj << 1)) * 256 + coll;
      lc = cprev[cb];
      rc = cprev[cb + 256];
    }
    float cn = tanhf(av) * sigm(iv) + sigm(f1) * lc + sigm(f2) * rc;
    float hn = sigm(ov) * tanhf(cn);
    long oidx = (long)((bb << log2n) + jj) * 256 + coll;
    cout[oidx] = cn;
    hout[oidx] = (bf16)hn;
  }
}

// ---------------- GCN GEMM1: t1T[b,s,m] = (g1 @ gc2_W)[m,s], g1 built on-the-fly (rank-1 + relu) ----------------
__global__ __launch_bounds__(256)
void k_gcn1(const float* __restrict__ u, const float* __restrict__ gc1W, const float* __restrict__ gc1b,
            const bf16* __restrict__ gc2T /*(256,512)*/, bf16* __restrict__ t1T /*(B,256,128)*/) {
  const int lane = threadIdx.x & 31;
  const int wave = blockIdx.x * 8 + (threadIdx.x >> 5);
  const int b = wave / 112;
  const int rr = wave - b * 112;
  const int m0 = (rr % 7) << 4;     // node tile (112 rows)
  const int s0 = (rr / 7) << 4;     // feature tile (256)
  const float uv = u[b * 112 + m0 + (lane & 15)];
  const bf16* colb = gc2T + (long)(s0 + (lane & 15)) * 512;
  v8f acc = v8f_zero();
  for (int kk = 0; kk < 512; kk += 32) {
    FragU f;
    const int kb = kk + ((lane >> 4) << 3);
    #pragma unroll
    for (int e = 0; e < 8; e++) { float g = fmaxf(uv * gc1W[kb + e] + gc1b[kb + e], 0.f); f.e[e] = (bf16)g; }
    #pragma unroll
    for (int e = 0; e < 8; e++) { int k = kb + 16 + e; float g = fmaxf(uv * gc1W[k] + gc1b[k], 0.f); f.e[8 + e] = (bf16)g; }
    v16bf bw = load_frag_b(colb, kk, lane);
    acc = wmma_bf16(f.v, bw, acc);
  }
  const int hi = (lane >> 4) << 3;
  PackU pk;
  #pragma unroll
  for (int r = 0; r < 8; r++) {
    int mm = m0 + hi + r;
    pk.e[r] = (bf16)((mm < 108) ? acc[r] : 0.f);
  }
  const int s = s0 + (lane & 15);
  *(u32x4*)(t1T + ((long)(b * 256 + s) * 128 + m0 + hi)) = pk.q;
}

// ---------------- GCN GEMM2: relu(adj@t1 + gc2_b), reduced by gcn_fc_W into tree_enc via atomics ----------------
__global__ __launch_bounds__(256)
void k_gcn2(const bf16* __restrict__ adjbf /*(B,112,128)*/, const bf16* __restrict__ t1T /*(B,256,128)*/,
            const float* __restrict__ gc2b, const float* __restrict__ gcnW,
            float* __restrict__ treeenc /*(B,256)*/) {
  const int lane = threadIdx.x & 31;
  const int wave = blockIdx.x * 8 + (threadIdx.x >> 5);
  const int b = wave / 112;
  const int rr = wave - b * 112;
  const int m0 = (rr % 7) << 4;
  const int s0 = (rr / 7) << 4;
  const bf16* arow = adjbf + ((long)b * 112 + m0 + (lane & 15)) * 128;
  const bf16* bcol = t1T + ((long)b * 256 + s0 + (lane & 15)) * 128;
  v8f acc = v8f_zero();
  #pragma unroll
  for (int kk = 0; kk < 128; kk += 32) {
    v16bf a = load_frag_a(arow, kk, lane);
    v16bf w = load_frag_b(bcol, kk, lane);
    acc = wmma_bf16(a, w, acc);
  }
  const int s = s0 + (lane & 15);
  const float bb = gc2b[s];
  const int hi = (lane >> 4) << 3;
  float part = 0.f;
  #pragma unroll
  for (int r = 0; r < 8; r++) {
    int m = m0 + hi + r;
    float w = (m < 108) ? gcnW[m] : 0.f;
    part += w * fmaxf(acc[r] + bb, 0.f);
  }
  atomicAdd(&treeenc[b * 256 + s], part);
}

// ---------------- concat [state|act|tree] -> xcat bf16 ----------------
__global__ __launch_bounds__(256)
void k_xcat(const bf16* __restrict__ hfin, const bf16* __restrict__ actenc,
            const float* __restrict__ tree, const float* __restrict__ gfcb, bf16* __restrict__ xcat) {
  int b = blockIdx.x, s = threadIdx.x;
  xcat[(long)b * 768 + s]        = hfin[(long)b * 256 + s];
  xcat[(long)b * 768 + 256 + s]  = actenc[(long)b * 256 + s];
  xcat[(long)b * 768 + 512 + s]  = (bf16)(tree[(long)b * 256 + s] + gfcb[0]);
}
// ---------------- final dot: out[b] = hidden[b,:]·out_W + out_b ----------------
__global__ __launch_bounds__(256)
void k_out(const float* __restrict__ hidden, const float* __restrict__ oW,
           const float* __restrict__ ob, float* __restrict__ out) {
  int b = blockIdx.x, t = threadIdx.x;
  float p = hidden[(long)b * 256 + t] * oW[t];
  __shared__ float red[8];
  #pragma unroll
  for (int o = 16; o > 0; o >>= 1) p += __shfl_down(p, o, 32);
  if ((t & 31) == 0) red[t >> 5] = p;
  __syncthreads();
  if (t == 0) {
    float s = 0.f;
    for (int w = 0; w < 8; w++) s += red[w];
    out[b] = s + ob[0];
  }
}

extern "C" void kernel_launch(void* const* d_in, const int* in_sizes, int n_in,
                              void* d_out, int out_size, void* d_ws, size_t ws_size,
                              hipStream_t stream) {
  const int*   leaf  = (const int*)  d_in[0];
  const int*   act   = (const int*)  d_in[1];
  const float* jcols = (const float*)d_in[2];
  const float* adj   = (const float*)d_in[3];
  const float* jvec  = (const float*)d_in[4];
  const float* emb   = (const float*)d_in[5];
  const float* ln_g  = (const float*)d_in[6];
  const float* ln_b  = (const float*)d_in[7];
  const float* jcW   = (const float*)d_in[8];
  const float* jcb   = (const float*)d_in[9];
  const float* fc0W  = (const float*)d_in[10];
  const float* fc0b  = (const float*)d_in[11];
  const float* fc1W  = (const float*)d_in[12];
  const float* fc1b  = (const float*)d_in[13];
  const float* fc2W  = (const float*)d_in[14];
  const float* fc2b  = (const float*)d_in[15];
  const float* gc1W  = (const float*)d_in[16];
  const float* gc1b  = (const float*)d_in[17];
  const float* gc2W  = (const float*)d_in[18];
  const float* gc2b  = (const float*)d_in[19];
  const float* gfcW  = (const float*)d_in[20];
  const float* gfcb  = (const float*)d_in[21];
  const float* fcW   = (const float*)d_in[22];
  const float* fcb   = (const float*)d_in[23];
  const float* outW  = (const float*)d_in[24];
  const float* outb  = (const float*)d_in[25];
  float* out = (float*)d_out;

  char* p = (char*)d_ws;
  auto take = [&](size_t bytes) -> char* { char* r = p; p += (bytes + 255) & ~((size_t)255); return r; };
  bf16*  wFC0   = (bf16*) take(1280L * 256 * 2);
  bf16*  wFC1   = (bf16*) take(1280L * 256 * 2);
  bf16*  wFC2   = (bf16*) take(1280L * 256 * 2);
  bf16*  wJC    = (bf16*) take(256L * 224 * 2);
  bf16*  wGC2T  = (bf16*) take(256L * 512 * 2);
  bf16*  wFCW   = (bf16*) take(256L * 768 * 2);
  bf16*  hA     = (bf16*) take(512L * 16 * 256 * 2);
  bf16*  hB     = (bf16*) take(512L * 8 * 256 * 2);
  float* cA     = (float*)take(512L * 8 * 256 * 4);
  float* cB     = (float*)take(512L * 4 * 256 * 4);
  bf16*  xpad   = (bf16*) take(512L * 15 * 224 * 2);
  bf16*  inputx = (bf16*) take(512L * 15 * 256 * 2);
  bf16*  adjbf  = (bf16*) take(512L * 112 * 128 * 2);
  float* ubuf   = (float*)take(512L * 112 * 4);
  bf16*  t1T    = (bf16*) take(512L * 256 * 128 * 2);
  float* tree   = (float*)take(512L * 256 * 4);
  bf16*  actenc = (bf16*) take(512L * 256 * 2);
  bf16*  xcat   = (bf16*) take(512L * 768 * 2);
  float* hidden = (float*)take(512L * 256 * 4);

  // zeros (t1T padding + tree_enc atomic target)
  k_zero32<<<2048, 256, 0, stream>>>((uint32_t*)t1T, 512L * 256 * 128 / 2);
  k_zero32<<<128, 256, 0, stream>>>((uint32_t*)tree, 512L * 256);
  // weight prep
  k_cvt_bf16<<<512, 256, 0, stream>>>(fc0W, wFC0, 1280L * 256);
  k_cvt_bf16<<<512, 256, 0, stream>>>(fc1W, wFC1, 1280L * 256);
  k_cvt_bf16<<<512, 256, 0, stream>>>(fc2W, wFC2, 1280L * 256);
  k_cvt_bf16<<<512, 256, 0, stream>>>(fcW,  wFCW, 256L * 768);
  k_padmat<<<128, 256, 0, stream>>>(jcW, wJC, 200, 224, 256L * 224);
  k_transpose_bf16<<<256, 256, 0, stream>>>(gc2W, wGC2T, 512, 256, 512L * 256);
  // embeddings + LN
  k_embed_ln<<<512 * 17, 256, 0, stream>>>(leaf, act, emb, ln_g, ln_b, hA, actenc);
  // join_cols -> padded bf16, then inputx GEMM (M=7680, K=224, N=256)
  k_padmat<<<2048, 256, 0, stream>>>(jcols, xpad, 200, 224, 7680L * 224);
  k_gemm256<<<960, 256, 0, stream>>>(xpad, wJC, jcb, nullptr, inputx, 224, 0, 480);
  // Tree-LSTM levels (ping-pong)
  struct Lv { const bf16* hp; const float* cp; bf16* ho; float* co; int n; int l2; int off; };
  Lv lv[4] = {
    { hA, nullptr, hB, cA, 8, 3, 0  },
    { hB, cA,      hA, cB, 4, 2, 8  },
    { hA, cB,      hB, cA, 2, 1, 12 },
    { hB, cA,      hA, cB, 1, 0, 14 },
  };
  for (int i = 0; i < 4; i++) {
    int tiles_m = (512 * lv[i].n) / 16;
    int blocks = (512 * lv[i].n) / 8;   // one wave per 16x16 tile, 5 gate tiles per wave
    k_tree_level<<<blocks, 256, 0, stream>>>(lv[i].hp, lv[i].cp, inputx,
                                             wFC1, wFC2, wFC0, fc1b, fc2b, fc0b,
                                             lv[i].ho, lv[i].co, lv[i].n, lv[i].l2, lv[i].off, tiles_m);
  }
  // GCN
  k_adjprep<<<4096, 256, 0, stream>>>(adj, adjbf, 512L * 112 * 128);
  k_u<<<512, 128, 0, stream>>>(adj, jvec, ubuf);
  k_gcn1<<<7168, 256, 0, stream>>>(ubuf, gc1W, gc1b, wGC2T, t1T);
  k_gcn2<<<7168, 256, 0, stream>>>(adjbf, t1T, gc2b, gfcW, tree);
  // final head
  k_xcat<<<512, 256, 0, stream>>>(hA, actenc, tree, gfcb, xcat);
  k_gemm256<<<64, 256, 0, stream>>>(xcat, wFCW, fcb, hidden, nullptr, 768, 1, 32);
  k_out<<<512, 256, 0, stream>>>(hidden, outW, outb, out);
}
